// TstudentSupervisedContrastiveLoss_31645319036929
// MI455X (gfx1250) — compile-verified, matrix operand order checked by reading, not run on hardware
//
#include <hip/hip_runtime.h>

typedef _Float16 v16h __attribute__((ext_vector_type(16)));
typedef _Float16 h8   __attribute__((ext_vector_type(8)));
typedef float    v8f  __attribute__((ext_vector_type(8)));

constexpr int NROWS = 8192;
constexpr int D     = 768;
constexpr int MB_ROWS = 64;                    // M rows per workgroup
constexpr int WAVES   = 4;                     // 4 waves x 16 rows
constexpr int NSPLIT  = 4;                     // N splits across blockIdx.y
constexpr int NCOLS_PER_SPLIT = NROWS / NSPLIT;
constexpr int NTILE   = 64;                    // 4 WMMA tiles per wave iteration

// ---------------------------------------------------------------- init
__global__ void tsc_init_kernel(float* __restrict__ denom, float* __restrict__ out) {
  int i = blockIdx.x * blockDim.x + threadIdx.x;
  if (i < NROWS) denom[i] = 0.0f;
  if (i == 0) out[0] = 0.0f;
}

// ------------------------------------------------------- normalize rows
__global__ __launch_bounds__(256) void tsc_normalize_kernel(
    const float* __restrict__ z1, const float* __restrict__ z2,
    _Float16* __restrict__ nh) {
  const int row = blockIdx.x;
  const float* src = (row < NROWS) ? (z1 + (size_t)row * D)
                                   : (z2 + (size_t)(row - NROWS) * D);
  _Float16* dst = nh + (size_t)row * D;
  const int tid = threadIdx.x;

  float v0 = src[tid], v1 = src[tid + 256], v2 = src[tid + 512];
  float ss = v0 * v0 + v1 * v1 + v2 * v2;
  for (int m = 16; m; m >>= 1) ss += __shfl_xor(ss, m, 32);

  __shared__ float red[8];
  if ((tid & 31) == 0) red[tid >> 5] = ss;
  __syncthreads();
  float tot = 0.0f;
  #pragma unroll
  for (int i = 0; i < 8; ++i) tot += red[i];

  const float s = 1.0f / fmaxf(sqrtf(tot), 1e-8f);   // torch CosineSimilarity clamp
  dst[tid]       = (_Float16)(v0 * s);
  dst[tid + 256] = (_Float16)(v1 * s);
  dst[tid + 512] = (_Float16)(v2 * s);
}

// -------------------------------------------- fused GEMM + t-Student epilogue
__global__ __launch_bounds__(WAVES * 32) void tsc_pairwise_kernel(
    const _Float16* __restrict__ n1h, const _Float16* __restrict__ n2h,
    float* __restrict__ denom, float* __restrict__ num) {
  __shared__ _Float16 Alds[MB_ROWS * D];   // 96 KB: this WG's A block

  const int tid  = threadIdx.x;
  const int wave = tid >> 5;
  const int lane = tid & 31;
  const int lg   = lane & 15;
  const bool hi  = (lane >= 16);
  const int mBase = blockIdx.x * MB_ROWS;

  // cooperative stage of A block (contiguous rows) into LDS, 16B chunks
  {
    const float4* gsrc = (const float4*)(n1h + (size_t)mBase * D);
    float4* ldst = (float4*)Alds;
    const int chunks = MB_ROWS * D / 8;  // 8 halves per float4
    for (int i = tid; i < chunks; i += WAVES * 32) ldst[i] = gsrc[i];
  }
  __syncthreads();

  const int m0 = mBase + wave * 16;
  const int kA = hi ? 8 : 0;             // K phase per 16-bit WMMA A layout
  const _Float16* aRow = Alds + (size_t)(wave * 16 + lg) * D;

  const int nBeg = blockIdx.y * NCOLS_PER_SPLIT;
  const int nEnd = nBeg + NCOLS_PER_SPLIT;

  float dsum[8];
  #pragma unroll
  for (int r = 0; r < 8; ++r) dsum[r] = 0.0f;

  for (int nBase = nBeg; nBase < nEnd; nBase += NTILE) {
    // warm L2/WGP$ for the next N block's B rows
    const int nNext = nBase + NTILE;
    if (nNext < nEnd) {
      #pragma unroll
      for (int j = 0; j < 4; ++j)
        __builtin_prefetch(n2h + (size_t)(nNext + j * 16 + lg) * D + kA, 0, 1);
    }

    const _Float16* bRow[4];
    #pragma unroll
    for (int j = 0; j < 4; ++j)
      bRow[j] = n2h + (size_t)(nBase + j * 16 + lg) * D + kA;

    v8f acc[4] = {};

    for (int k0 = 0; k0 < D; k0 += 32) {
      union { v16h v; h8 h[2]; } a;
      a.h[0] = *(const h8*)(aRow + k0 + kA);        // K {k0..k0+7}(+phase)
      a.h[1] = *(const h8*)(aRow + k0 + kA + 16);   // K {k0+16..k0+23}(+phase)
      #pragma unroll
      for (int j = 0; j < 4; ++j) {
        union { v16h v; h8 h[2]; } b;
        b.h[0] = *(const h8*)(bRow[j] + k0);
        b.h[1] = *(const h8*)(bRow[j] + k0 + 16);
        acc[j] = __builtin_amdgcn_wmma_f32_16x16x32_f16(
            false, a.v, false, b.v, (short)0, acc[j], false, false);
      }
    }

    // fused epilogue: f(c) = (1.5 - 0.5c)^(-3/2) = rsqrt(t)^3
    #pragma unroll
    for (int j = 0; j < 4; ++j) {
      const int n0 = nBase + j * 16;
      const bool diagTile = (n0 == m0);
      #pragma unroll
      for (int r = 0; r < 8; ++r) {
        const float c  = acc[j][r];
        const float t  = fmaf(-0.5f, c, 1.5f);
        const float rs = rsqrtf(t);
        const float f  = rs * rs * rs;
        dsum[r] += f;
        const int mloc = r + (hi ? 8 : 0);          // C layout: VGPR r -> M, lane -> N
        if (diagTile && mloc == lg) num[m0 + mloc] = f;
      }
    }
  }

  // reduce across the 16 lanes holding the tile's N columns (per half-wave)
  #pragma unroll
  for (int r = 0; r < 8; ++r) {
    float v = dsum[r];
    v += __shfl_xor(v, 1, 32);
    v += __shfl_xor(v, 2, 32);
    v += __shfl_xor(v, 4, 32);
    v += __shfl_xor(v, 8, 32);
    if (lg == 0) atomicAdd(&denom[m0 + r + (hi ? 8 : 0)], v);
  }
}

// ---------------------------------------------------------------- finalize
__global__ __launch_bounds__(256) void tsc_finalize_kernel(
    const float* __restrict__ num, const float* __restrict__ denom,
    float* __restrict__ out) {
  const int i = blockIdx.x * 256 + threadIdx.x;
  float p = num[i] / denom[i];
  for (int m = 16; m; m >>= 1) p += __shfl_xor(p, m, 32);
  __shared__ float red[8];
  if ((threadIdx.x & 31) == 0) red[threadIdx.x >> 5] = p;
  __syncthreads();
  if (threadIdx.x == 0) {
    float t = 0.0f;
    #pragma unroll
    for (int k = 0; k < 8; ++k) t += red[k];
    atomicAdd(out, -t / (float)NROWS);
  }
}

extern "C" void kernel_launch(void* const* d_in, const int* in_sizes, int n_in,
                              void* d_out, int out_size, void* d_ws, size_t ws_size,
                              hipStream_t stream) {
  const float* z1 = (const float*)d_in[0];
  const float* z2 = (const float*)d_in[1];
  float* out = (float*)d_out;

  _Float16* nh = (_Float16*)d_ws;                          // [2*8192][768] f16
  const size_t nhBytes = (size_t)2 * NROWS * D * sizeof(_Float16);
  float* denom = (float*)((char*)d_ws + nhBytes);          // [8192] f32
  float* num   = denom + NROWS;                            // [8192] f32

  tsc_init_kernel<<<(NROWS + 255) / 256, 256, 0, stream>>>(denom, out);
  tsc_normalize_kernel<<<2 * NROWS, 256, 0, stream>>>(z1, z2, nh);
  dim3 grid(NROWS / MB_ROWS, NSPLIT);                      // 128 x 4 workgroups
  tsc_pairwise_kernel<<<grid, WAVES * 32, 0, stream>>>(nh, nh + (size_t)NROWS * D,
                                                       denom, num);
  tsc_finalize_kernel<<<NROWS / 256, 256, 0, stream>>>(num, denom, out);
}
